// HL_LH_block_1_78151224918206
// MI455X (gfx1250) — matrix operand order repaired and checked by reading, not compiled, not run on hardware
//
#include <hip/hip_runtime.h>

// ---------------------------------------------------------------------------
// CDNA5 / gfx1250 fused EGAT pipeline.
// Dense parts use v_wmma_f32_16x16x32_bf16 (edges-as-M for the per-edge GEMM);
// sparse parts are gather + atomic scatter (global_atomic_add_f32).
// ---------------------------------------------------------------------------

typedef __bf16 bf16_t;
typedef bf16_t v16bf __attribute__((ext_vector_type(16)));
typedef float  v8f   __attribute__((ext_vector_type(8)));

#define N_GG 50000
#define E_GG 800000
#define N_LG 40000
#define E_LG 100000
#define HEADS 8

// Native conversion: lowers to v_cvt_pk_bf16_f32 pairs (RNE).
__device__ __forceinline__ void pack8(v16bf& v, int base, float4 a, float4 b) {
  v[base + 0] = (bf16_t)a.x; v[base + 1] = (bf16_t)a.y;
  v[base + 2] = (bf16_t)a.z; v[base + 3] = (bf16_t)a.w;
  v[base + 4] = (bf16_t)b.x; v[base + 5] = (bf16_t)b.y;
  v[base + 6] = (bf16_t)b.z; v[base + 7] = (bf16_t)b.w;
}

// A-matrix 16x32 bf16 WMMA layout (ISA 7.12.2):
// lane&15 = row M; g=lane>>4: v0..v3 hold K=8g+0..7 (2/VGPR), v4..v7 hold K=16+8g+0..7
// All fragment bases are 32B-aligned -> float4 loads become global_load_b128.
__device__ __forceinline__ v16bf load_A16x32(const float* __restrict__ A,
                                             long row, int lda, int kb, int g) {
  const float* base = A + row * (long)lda + kb;
  const float4* p = reinterpret_cast<const float4*>(base + 8 * g);
  const float4* q = reinterpret_cast<const float4*>(base + 16 + 8 * g);
  v16bf a;
  pack8(a, 0, p[0], p[1]);
  pack8(a, 8, q[0], q[1]);
  return a;
}

// B-matrix 32x16 bf16 WMMA layout: lane&15 = col N; half-wave g holds K=8g+0..7 then K=16+8g+0..7.
// W is [N, K] row-major (torch Linear weight), so B(k, n) = W[n*K + k].
__device__ __forceinline__ v16bf load_B32x16(const float* __restrict__ W,
                                             int n, int ldw, int kb, int g) {
  const float4* p = reinterpret_cast<const float4*>(W + (long)n * ldw + kb + 8 * g);
  const float4* q = reinterpret_cast<const float4*>(W + (long)n * ldw + kb + 16 + 8 * g);
  v16bf b;
  pack8(b, 0, p[0], p[1]);
  pack8(b, 8, q[0], q[1]);
  return b;
}

__device__ __forceinline__ v8f wmma_bf16(v16bf a, v16bf b, v8f c) {
  return __builtin_amdgcn_wmma_f32_16x16x32_bf16(false, a, false, b,
                                                 (short)0, c, false, false);
}

// ---------------------------------------------------------------------------
// Generic C = A @ W^T (+bias). A:[M,K] f32, W:[N,K] f32, C:[M,N] f32.
// grid = (M/16, N/16), block = 32 (one wave per 16x16 tile). K is static.
// ---------------------------------------------------------------------------
template <int K>
__global__ void gemm_wmma(const float* __restrict__ A, const float* __restrict__ W,
                          const float* __restrict__ bias, float* __restrict__ C,
                          int N) {
  int lane = threadIdx.x;
  int l = lane & 15, g = lane >> 4;
  long row16 = (long)blockIdx.x * 16;
  int n = blockIdx.y * 16 + l;
  v8f acc = {0.f, 0.f, 0.f, 0.f, 0.f, 0.f, 0.f, 0.f};
#pragma unroll
  for (int c = 0; c < K / 32; ++c) {
    v16bf a = load_A16x32(A, row16 + l, K, 32 * c, g);
    v16bf b = load_B32x16(W, n, K, 32 * c, g);
    acc = wmma_bf16(a, b, acc);
  }
  float bv = bias ? bias[n] : 0.0f;
  // D layout: lane -> col n; VGPR r -> row r + 8*g
#pragma unroll
  for (int r = 0; r < 8; ++r) {
    long m = row16 + r + 8 * g;
    C[m * (long)N + n] = acc[r] + bv;
  }
}

// ---------------------------------------------------------------------------
// Fused per-edge kernel: 16 edges per wave (edges are M of the WMMA GEMM).
//   f = (ef @ Wfij^T) + ni[src] + nj[dst] + bias ; leaky_relu
//   Eout[e,h]  = sum_fe f * attn      (cross-lane shfl reduce per head)
//   EFsum[e,:] = sum_h  f             (LDS tile accumulate; optional)
// N = HEADS*Fe is compile-time; K in {32,64}; Fe in {32,64}.
// ---------------------------------------------------------------------------
template <int K, int Fe>
__global__ void edge_fused(const float* __restrict__ EF,
                           const float* __restrict__ Wfij,
                           const float* __restrict__ bias,
                           const float* __restrict__ NI,
                           const float* __restrict__ NJ,
                           const float* __restrict__ attn,
                           const int* __restrict__ src,
                           const int* __restrict__ dst,
                           float* __restrict__ Eout,
                           float* __restrict__ EFsum) {
  constexpr int N = HEADS * Fe;
  constexpr int headTiles = Fe / 16;   // 16-col tiles per head
  constexpr int ntiles = N / 16;
  __shared__ int s_src[16];
  __shared__ int s_dst[16];
  __shared__ float s_ef[16][64];
  int lane = threadIdx.x;
  int l = lane & 15, g = lane >> 4;
  long e0 = (long)blockIdx.x * 16;
  if (g == 0) s_src[l] = src[e0 + l];
  else        s_dst[l] = dst[e0 + l];
  if (EFsum) {
    for (int i = lane; i < 16 * 64; i += 32) (&s_ef[0][0])[i] = 0.0f;
  }
  __syncthreads();

  v16bf aPack[K / 32];
#pragma unroll
  for (int c = 0; c < K / 32; ++c)
    aPack[c] = load_A16x32(EF, e0 + l, K, 32 * c, g);

  float eacc[8];
#pragma unroll
  for (int r = 0; r < 8; ++r) eacc[r] = 0.0f;

  for (int nt = 0; nt < ntiles; ++nt) {
    int n = nt * 16 + l;
    v8f d = {0.f, 0.f, 0.f, 0.f, 0.f, 0.f, 0.f, 0.f};
#pragma unroll
    for (int c = 0; c < K / 32; ++c)
      d = wmma_bf16(aPack[c], load_B32x16(Wfij, n, K, 32 * c, g), d);
    float an = attn[n];   // attn flat [HEADS*Fe] == index n
    float bn = bias[n];
    int fe = n & (Fe - 1);
#pragma unroll
    for (int r = 0; r < 8; ++r) {
      int m = r + 8 * g;                 // edge-in-tile this D element belongs to
      long si = s_src[m];
      long di = s_dst[m];
      float f = d[r] + NI[si * (long)N + n] + NJ[di * (long)N + n] + bn;
      f = (f > 0.0f) ? f : 0.01f * f;    // leaky_relu(0.01)
      eacc[r] += f * an;
      if (EFsum) s_ef[m][fe] += f;       // same lane owns (m,fe) each visit: no race
    }
    if ((nt & (headTiles - 1)) == (headTiles - 1)) {
      int head = nt / headTiles;
#pragma unroll
      for (int r = 0; r < 8; ++r) {
        float v = eacc[r];
        v += __shfl_xor(v, 1, 32);
        v += __shfl_xor(v, 2, 32);
        v += __shfl_xor(v, 4, 32);
        v += __shfl_xor(v, 8, 32);       // reduce within each 16-lane half
        if (l == 0) Eout[(e0 + r + 8 * g) * HEADS + head] = v;
        eacc[r] = 0.0f;
      }
    }
  }
  if (EFsum) {
    __syncthreads();
    for (int i = lane; i < 16 * Fe; i += 32) {
      int m = i / Fe;
      int fe = i - m * Fe;
      EFsum[(e0 + m) * (long)Fe + fe] = s_ef[m][fe];
    }
  }
}

// ---------------------------------------------------------------------------
// Segment softmax passes (atomics) and weighted scatter.
// ---------------------------------------------------------------------------
__device__ __forceinline__ void atomic_max_f32(float* addr, float val) {
  int* ai = (int*)addr;
  int old = *(volatile int*)ai;
  while (__builtin_bit_cast(float, old) < val) {
    int assumed = old;
    old = atomicCAS(ai, assumed, __builtin_bit_cast(int, val));
    if (old == assumed) break;
  }
}

__global__ void seg_max_k(const float* __restrict__ e, const int* __restrict__ dst,
                          float* __restrict__ emax, long EH) {
  long i = (long)blockIdx.x * blockDim.x + threadIdx.x;
  if (i >= EH) return;
  long edge = i >> 3;
  int h = (int)(i & 7);
  atomic_max_f32(&emax[(long)dst[edge] * HEADS + h], e[i]);
}

__global__ void exp_den_k(float* __restrict__ e, const int* __restrict__ dst,
                          const float* __restrict__ emax, float* __restrict__ den,
                          long EH) {
  long i = (long)blockIdx.x * blockDim.x + threadIdx.x;
  if (i >= EH) return;
  long edge = i >> 3;
  int h = (int)(i & 7);
  long o = (long)dst[edge] * HEADS + h;
  float ex = __expf(e[i] - emax[o]);
  e[i] = ex;
  atomicAdd(&den[o], ex);
}

// out[dst, fn] += sum_h (ex/den) * Hf[src, h*64+fn]   (head-sum folded in:
// 8x fewer atomics). Hf stride fixed 512, Fn=64, H=8. 8 edges per 256-thr block.
__global__ void scatter_k(const float* __restrict__ ex, const float* __restrict__ den,
                          const int* __restrict__ src, const int* __restrict__ dst,
                          const float* __restrict__ Hf, float* __restrict__ out,
                          long E) {
  long edge = (long)blockIdx.x * 8 + (threadIdx.x >> 5);
  if (edge >= E) return;
  int lane = threadIdx.x & 31;
  long s = src[edge];
  long d = dst[edge];
  float a[HEADS];
#pragma unroll
  for (int h = 0; h < HEADS; ++h)
    a[h] = ex[edge * HEADS + h] / den[d * HEADS + h];
#pragma unroll
  for (int j = 0; j < 2; ++j) {
    int fn = lane + 32 * j;
    float sum = 0.0f;
#pragma unroll
    for (int h = 0; h < HEADS; ++h)
      sum += a[h] * Hf[s * 512 + h * 64 + fn];
    atomicAdd(&out[d * 64 + fn], sum);
  }
}

__global__ void fill_f32(float* __restrict__ p, float v, long n) {
  long i = (long)blockIdx.x * blockDim.x + threadIdx.x;
  long stride = (long)gridDim.x * blockDim.x;
  for (; i < n; i += stride) p[i] = v;
}

// ef_lg[i, :] = node_path[i/2, :]  (jnp.repeat(.., 2, axis=0))
__global__ void repeat2_k(const float* __restrict__ np_, float* __restrict__ ef, long n) {
  long i = (long)blockIdx.x * blockDim.x + threadIdx.x;
  long stride = (long)gridDim.x * blockDim.x;
  for (; i < n; i += stride) {
    long row = i >> 6;
    int c = (int)(i & 63);
    ef[i] = np_[(row >> 1) * 64 + c];
  }
}

// ---------------------------------------------------------------------------
extern "C" void kernel_launch(void* const* d_in, const int* in_sizes, int n_in,
                              void* d_out, int out_size, void* d_ws, size_t ws_size,
                              hipStream_t stream) {
  (void)in_sizes; (void)n_in; (void)out_size; (void)ws_size;

  const float* node_feats = (const float*)d_in[0];
  const float* node_path  = (const float*)d_in[2];
  const float* edge_path  = (const float*)d_in[3];
  const int*   gg_src     = (const int*)d_in[4];
  const int*   gg_dst     = (const int*)d_in[5];
  const int*   lg_src     = (const int*)d_in[6];
  const int*   lg_dst     = (const int*)d_in[7];
  const float* l1_Wnode   = (const float*)d_in[8];
  const float* l1_bnode   = (const float*)d_in[9];
  const float* l1_Wni     = (const float*)d_in[10];
  const float* l1_Wnj     = (const float*)d_in[11];
  const float* l1_Wfij    = (const float*)d_in[12];
  const float* l1_attn    = (const float*)d_in[13];
  const float* l1_bias    = (const float*)d_in[14];
  const float* g2_Wnode   = (const float*)d_in[15];
  const float* g2_bnode   = (const float*)d_in[16];
  const float* g2_Wni     = (const float*)d_in[17];
  const float* g2_Wnj     = (const float*)d_in[18];
  const float* g2_Wfij    = (const float*)d_in[19];
  const float* g2_attn    = (const float*)d_in[20];
  const float* g2_bias    = (const float*)d_in[21];

  // workspace layout (floats); buffers reused across stages
  float* ws    = (float*)d_ws;
  float* ni    = ws;                        // max(50000*256, 40000*512) = 20.48M
  float* nj    = ni + 20480000;             // 20.48M
  float* hb    = nj + 20480000;             // max(50000*512, 40000*512) = 25.6M
  float* ebuf  = hb + 25600000;             // max(E*H) = 6.4M
  float* emax  = ebuf + 6400000;            // 0.4M
  float* den   = emax + 400000;             // 0.4M
  float* npacc = den + 400000;              // 50000*64 = 3.2M
  float* ef0   = npacc + 3200000;           // 100000*64 = 6.4M
  float* ef1   = ef0 + 6400000;             // 6.4M
  float* nf1   = ef1 + 6400000;             // 40000*64 = 2.56M
  // total ~92.3M floats (~370 MB)

  float* outNF = (float*)d_out;             // [40000, 64]
  float* outEF = outNF + (long)N_LG * 64;   // [100000, 64]

  // ---------------- Stage 1: EGAT on gg ----------------
  fill_f32<<<2048, 256, 0, stream>>>(emax, -3.0e38f, (long)N_GG * HEADS);
  fill_f32<<<2048, 256, 0, stream>>>(den, 0.0f, (long)N_GG * HEADS);
  fill_f32<<<2048, 256, 0, stream>>>(npacc, 0.0f, (long)N_GG * 64);

  gemm_wmma<64><<<dim3(N_GG / 16, 256 / 16), 32, 0, stream>>>(node_path, l1_Wni, nullptr, ni, 256);
  gemm_wmma<64><<<dim3(N_GG / 16, 256 / 16), 32, 0, stream>>>(node_path, l1_Wnj, nullptr, nj, 256);
  gemm_wmma<64><<<dim3(N_GG / 16, 512 / 16), 32, 0, stream>>>(node_path, l1_Wnode, l1_bnode, hb, 512);

  edge_fused<32, 32><<<E_GG / 16, 32, 0, stream>>>(edge_path, l1_Wfij, l1_bias, ni, nj,
                                                   l1_attn, gg_src, gg_dst, ebuf, nullptr);
  {
    long EH = (long)E_GG * HEADS;
    int gb = (int)((EH + 255) / 256);
    seg_max_k<<<gb, 256, 0, stream>>>(ebuf, gg_dst, emax, EH);
    exp_den_k<<<gb, 256, 0, stream>>>(ebuf, gg_dst, emax, den, EH);
  }
  scatter_k<<<E_GG / 8, 256, 0, stream>>>(ebuf, den, gg_src, gg_dst, hb, npacc, (long)E_GG);
  repeat2_k<<<2048, 256, 0, stream>>>(npacc, ef0, (long)E_LG * 64);

  // ---------------- Stage 2: two stacked EGAT layers on lg ----------------
  const float* nf_in = node_feats;
  const float* ef_in = ef0;
  for (int L = 0; L < 2; ++L) {
    const float* Wnode = g2_Wnode + (long)L * 512 * 64;
    const float* bnode = g2_bnode + (long)L * 512;
    const float* Wni_  = g2_Wni   + (long)L * 512 * 64;
    const float* Wnj_  = g2_Wnj   + (long)L * 512 * 64;
    const float* Wfij_ = g2_Wfij  + (long)L * 512 * 64;
    const float* attn_ = g2_attn  + (long)L * 512;
    const float* bias_ = g2_bias  + (long)L * 512;
    float* nf_out = (L == 1) ? outNF : nf1;
    float* ef_out = (L == 1) ? outEF : ef1;

    fill_f32<<<2048, 256, 0, stream>>>(emax, -3.0e38f, (long)N_LG * HEADS);
    fill_f32<<<2048, 256, 0, stream>>>(den, 0.0f, (long)N_LG * HEADS);
    fill_f32<<<2048, 256, 0, stream>>>(nf_out, 0.0f, (long)N_LG * 64);

    gemm_wmma<64><<<dim3(N_LG / 16, 512 / 16), 32, 0, stream>>>(nf_in, Wni_, nullptr, ni, 512);
    gemm_wmma<64><<<dim3(N_LG / 16, 512 / 16), 32, 0, stream>>>(nf_in, Wnj_, nullptr, nj, 512);
    gemm_wmma<64><<<dim3(N_LG / 16, 512 / 16), 32, 0, stream>>>(nf_in, Wnode, bnode, hb, 512);

    edge_fused<64, 64><<<E_LG / 16, 32, 0, stream>>>(ef_in, Wfij_, bias_, ni, nj,
                                                     attn_, lg_src, lg_dst, ebuf, ef_out);
    {
      long EH = (long)E_LG * HEADS;
      int gb = (int)((EH + 255) / 256);
      seg_max_k<<<gb, 256, 0, stream>>>(ebuf, lg_dst, emax, EH);
      exp_den_k<<<gb, 256, 0, stream>>>(ebuf, lg_dst, emax, den, EH);
    }
    scatter_k<<<E_LG / 8, 256, 0, stream>>>(ebuf, den, lg_src, lg_dst, hb, nf_out, (long)E_LG);

    nf_in = nf_out;
    ef_in = ef_out;
  }
}
